// RelativeMultiHeadAttention_56435870269702
// MI455X (gfx1250) — matrix operand order repaired and checked by reading, not compiled, not run on hardware
//
#include <hip/hip_runtime.h>
#include <hip/hip_bf16.h>

// ---------------------------------------------------------------------------
// Transformer-XL Relative Multi-Head Attention for MI455X (gfx1250, wave32)
//   B=2, L=2048, D_MODEL=512, H=16, D_HEAD=32
// All GEMM-shaped math runs through V_WMMA_F32_16X16X32_BF16 (K=32 == d_head).
// Every WMMA operand is pre-packed to bf16 (u32 pairs) in operand-ready
// layout, so all hot loops are pure b128 loads + v_wmma.
// ---------------------------------------------------------------------------

#define D_MODEL 512
#define NUM_HEADS 16
#define D_HEAD 32
#define BATCH 2
#define L_SEQ 2048
#define SCALE 0.044194173824159216f   // 1/sqrt(512)

typedef __attribute__((ext_vector_type(16))) __bf16        v16bf;
typedef __attribute__((ext_vector_type(8)))  float         v8f;
typedef __attribute__((ext_vector_type(8)))  unsigned int  v8u;

// Pack two f32 into one VGPR holding two bf16 (RNE), merged with one V_PERM_B32.
__device__ __forceinline__ unsigned pk_bf16(float lo, float hi) {
  unsigned ulo = __float_as_uint(lo);
  unsigned uhi = __float_as_uint(hi);
  ulo += 0x7FFFu + ((ulo >> 16) & 1u);
  uhi += 0x7FFFu + ((uhi >> 16) & 1u);
  return __builtin_amdgcn_perm(uhi, ulo, 0x07060302u);
}

__device__ __forceinline__ v8u pk8(float4 a0, float4 a1, float4 a2, float4 a3) {
  v8u r;
  r[0] = pk_bf16(a0.x, a0.y); r[1] = pk_bf16(a0.z, a0.w);
  r[2] = pk_bf16(a1.x, a1.y); r[3] = pk_bf16(a1.z, a1.w);
  r[4] = pk_bf16(a2.x, a2.y); r[5] = pk_bf16(a2.z, a2.w);
  r[6] = pk_bf16(a3.x, a3.y); r[7] = pk_bf16(a3.z, a3.w);
  return r;
}

__device__ __forceinline__ float4 f4add(float4 a, float4 b) {
  return make_float4(a.x + b.x, a.y + b.y, a.z + b.z, a.w + b.w);
}

__device__ __forceinline__ v8u mk8(uint4 x, uint4 y) {  // two uint4 -> operand
  v8u r;
  r[0] = x.x; r[1] = x.y; r[2] = x.z; r[3] = x.w;
  r[4] = y.x; r[5] = y.y; r[6] = y.z; r[7] = y.w;
  return r;
}

__device__ __forceinline__ v8f wmma_pk(v8u a, v8u b, v8f c) {
  return __builtin_amdgcn_wmma_f32_16x16x32_bf16(
      false, __builtin_bit_cast(v16bf, a),
      false, __builtin_bit_cast(v16bf, b),
      (short)0, c, false, false);
}

// ---------------------------------------------------------------------------
// Kernel 0: generic f32 -> packed-bf16-pairs (row-major). dst[t]=pk(src[2t],src[2t+1])
// Used for query/key/value/pos and the five 512x512 weight matrices.
// ---------------------------------------------------------------------------
__global__ void pack_pairs_bf16(const float* __restrict__ src,
                                unsigned* __restrict__ dst, int n) {
  const int t = blockIdx.x * 256 + threadIdx.x;
  if (t >= n) return;
  const float2 p = ((const float2*)src)[t];
  dst[t] = pk_bf16(p.x, p.y);
}

// ---------------------------------------------------------------------------
// Kernel 1:  Y[M x 512] = X @ W^T + bias, X/W pre-packed bf16 pairs (256 u32/row)
// One 16x16 tile per wave; 8 waves / 256-thread block. grid = (M/16, 4).
// Inner loop: 4x global_load_b128 + 1x v_wmma, no conversion VALU.
// ---------------------------------------------------------------------------
__global__ void gemm_bias_wmma(const unsigned* __restrict__ Xp,
                               const unsigned* __restrict__ Wp,
                               const float* __restrict__ bias,
                               float* __restrict__ Y,
                               int M) {
  const int lane = threadIdx.x & 31;
  const int wave = threadIdx.x >> 5;
  const int half = lane >> 4;
  const int lr   = lane & 15;

  const int tm = blockIdx.x;
  const int tn = blockIdx.y * 8 + wave;

  const int row = tm * 16 + lr;
  const int col = tn * 16 + lr;
  const unsigned* xrow = Xp + (size_t)row * 256 + (half << 2);  // + half*4 pairs
  const unsigned* wrow = Wp + (size_t)col * 256 + (half << 2);  // + half*4 pairs

  const float bv = bias ? bias[col] : 0.0f;  // hoisted: same col for all 8 outputs

  v8f acc = {};
#pragma unroll
  for (int k0 = 0; k0 < 256; k0 += 16) {     // 16 u32 pairs == 32 K values
    const uint4* x4 = (const uint4*)(xrow + k0);
    const uint4* w4 = (const uint4*)(wrow + k0);
    const v8u a = mk8(x4[0], x4[2]);         // pair runs [0..3], [8..11] (+half*4)
    const v8u b = mk8(w4[0], w4[2]);         // pair runs [0..3], [8..11] (+half*4)
    acc = wmma_pk(a, b, acc);
  }

  const int base = (tm * 16 + 8 * half) * D_MODEL + col;
#pragma unroll
  for (int v = 0; v < 8; ++v)
    Y[base + v * D_MODEL] = acc[v] + bv;
}

// ---------------------------------------------------------------------------
// Kernel 2a: pack K or P rows to per-head bf16, row-major (bh, j, d) pairs.
// ---------------------------------------------------------------------------
__global__ void pack_rows_bf16(const float* __restrict__ src,
                               unsigned* __restrict__ dst, int batch) {
  const int t = blockIdx.x * 256 + threadIdx.x;
  const int total = batch * NUM_HEADS * L_SEQ * (D_HEAD / 2);
  if (t >= total) return;
  const int d2  = t & 15;
  const int ri  = t >> 4;
  const int j   = ri & (L_SEQ - 1);
  const int bh  = ri >> 11;                // / L_SEQ
  const int h   = bh & (NUM_HEADS - 1);
  const int b   = bh >> 4;
  const float2 p = *(const float2*)(src + ((size_t)(b * L_SEQ + j)) * D_MODEL
                                        + h * D_HEAD + (d2 << 1));
  dst[t] = pk_bf16(p.x, p.y);
}

// ---------------------------------------------------------------------------
// Kernel 2b: pack V pair-wise along j for the attn@V B-operand.
// ---------------------------------------------------------------------------
__global__ void pack_vpairs_bf16(const float* __restrict__ src,
                                 unsigned* __restrict__ dst) {
  const int t = blockIdx.x * 256 + threadIdx.x;
  const int total = BATCH * NUM_HEADS * (L_SEQ / 2) * D_HEAD;
  if (t >= total) return;
  const int d   = t & 31;
  const int k2  = (t >> 5) & (L_SEQ / 2 - 1);
  const int bh  = t >> 15;                 // / (32 * L/2)
  const int h   = bh & (NUM_HEADS - 1);
  const int b   = bh >> 4;
  const float* s0 = src + ((size_t)(b * L_SEQ + 2 * k2)) * D_MODEL + h * D_HEAD + d;
  dst[t] = pk_bf16(s0[0], s0[D_MODEL]);
}

// ---------------------------------------------------------------------------
// Kernel 3: relative attention core.
// grid = (B*H, L/16), block = 128 (4 waves). Dynamic LDS:
//   sP: 17x2048 f32 | sS: 16x2048 f32 | sC: 16x32 f32 | red: 128 f32
// Emits its context block directly as packed bf16 pairs for the Wo GEMM.
// ---------------------------------------------------------------------------
__global__ void rel_attn_wmma(const float* __restrict__ Qf,
                              const unsigned* __restrict__ Kb,
                              const unsigned* __restrict__ Vb,
                              const unsigned* __restrict__ Pb,
                              const float* __restrict__ u_bias,
                              const float* __restrict__ v_bias,
                              unsigned* __restrict__ ctxp) {
  extern __shared__ float lds[];
  float* sP  = lds;                        // 17*L_SEQ
  float* sS  = sP + 17 * L_SEQ;            // 16*L_SEQ
  float* sC  = sS + 16 * L_SEQ;            // 16*D_HEAD
  float* red = sC + 16 * D_HEAD;           // 128

  const int bh  = blockIdx.x;
  const int b   = bh >> 4;
  const int h   = bh & (NUM_HEADS - 1);
  const int i0  = blockIdx.y * 16;
  const int tid = threadIdx.x;
  const int lane = tid & 31;
  const int wave = tid >> 5;
  const int half = lane >> 4;
  const int lr   = lane & 15;

  // ---- Preload A operands (reused across all 128 j-tiles) ----------------
  v8u aQu, aQv0, aQv1;
  {
    const int r0 = i0 + lr;
    int r1 = i0 + 16 + lr; if (r1 >= L_SEQ) r1 = L_SEQ - 1;   // clamp, unused rows
    const float4* q0 = (const float4*)(Qf + ((size_t)(b * L_SEQ + r0)) * D_MODEL + h * D_HEAD + (half << 3));
    const float4* q1 = (const float4*)(Qf + ((size_t)(b * L_SEQ + r1)) * D_MODEL + h * D_HEAD + (half << 3));
    const float4* u4 = (const float4*)(u_bias + h * D_HEAD + (half << 3));
    const float4* v4 = (const float4*)(v_bias + h * D_HEAD + (half << 3));
    float4 qa0 = q0[0], qa1 = q0[1], qa2 = q0[4], qa3 = q0[5];
    float4 qb0 = q1[0], qb1 = q1[1], qb2 = q1[4], qb3 = q1[5];
    float4 ua0 = u4[0], ua1 = u4[1], ua2 = u4[4], ua3 = u4[5];
    float4 va0 = v4[0], va1 = v4[1], va2 = v4[4], va3 = v4[5];
    aQu  = pk8(f4add(qa0, ua0), f4add(qa1, ua1), f4add(qa2, ua2), f4add(qa3, ua3));
    aQv0 = pk8(f4add(qa0, va0), f4add(qa1, va1), f4add(qa2, va2), f4add(qa3, va3));
    aQv1 = pk8(f4add(qb0, va0), f4add(qb1, va1), f4add(qb2, va2), f4add(qb3, va3));
  }

  // ---- Stage 1: position scores sP[r][j] = (Q+v)[i0+r] . P[j], r=0..16 ----
  for (int jt = wave; jt < L_SEQ / 16; jt += 4) {
    const int n = jt * 16 + lr;
    const uint4* p4 = (const uint4*)(Pb + ((size_t)(h * L_SEQ + n)) * 16 + (half << 2));
    const v8u bP = mk8(p4[0], p4[2]);
    v8f c0 = {}, c1 = {};
    c0 = wmma_pk(aQv0, bP, c0);        // rows i0..i0+15
    c1 = wmma_pk(aQv1, bP, c1);        // rows i0+16..i0+31 (only row 16 kept)
#pragma unroll
    for (int v = 0; v < 8; ++v) {
      const int r = v + 8 * half;
      sP[r * L_SEQ + jt * 16 + lr] = c0[v];
    }
    if (half == 0)
      sP[16 * L_SEQ + jt * 16 + lr] = c1[0];
  }
  __syncthreads();

  // ---- Stage 2: content scores + relative shift -> sS ---------------------
  for (int jt = wave; jt < L_SEQ / 16; jt += 4) {
    const int n = jt * 16 + lr;
    const uint4* k4 = (const uint4*)(Kb + ((size_t)(bh * L_SEQ + n)) * 16 + (half << 2));
    const v8u bK = mk8(k4[0], k4[2]);
    v8f c = {};
    c = wmma_pk(aQu, bK, c);
#pragma unroll
    for (int v = 0; v < 8; ++v) {
      const int r = v + 8 * half;
      const int i = i0 + r;
      const int j = jt * 16 + lr;
      // Transformer-XL pad/reshape shift, including un-masked wrap for j>i+1
      float pos;
      if (j <= i)          pos = sP[r * L_SEQ + (L_SEQ - 1 + j - i)];
      else if (j == i + 1) pos = 0.0f;
      else                 pos = sP[(r + 1) * L_SEQ + (j - i - 2)];
      sS[r * L_SEQ + j] = (c[v] + pos) * SCALE;
    }
  }
  __syncthreads();

  // ---- Stage 3: row softmax over sS (16 rows x 2048) ----------------------
  for (int r = 0; r < 16; ++r) {
    float mx = -3.0e38f;
    for (int j = tid; j < L_SEQ; j += 128) mx = fmaxf(mx, sS[r * L_SEQ + j]);
    red[tid] = mx; __syncthreads();
    for (int s = 64; s > 0; s >>= 1) {
      if (tid < s) red[tid] = fmaxf(red[tid], red[tid + s]);
      __syncthreads();
    }
    mx = red[0]; __syncthreads();

    float sum = 0.0f;
    for (int j = tid; j < L_SEQ; j += 128) {
      float e = __expf(sS[r * L_SEQ + j] - mx);
      sS[r * L_SEQ + j] = e;
      sum += e;
    }
    red[tid] = sum; __syncthreads();
    for (int s = 64; s > 0; s >>= 1) {
      if (tid < s) red[tid] += red[tid + s];
      __syncthreads();
    }
    const float inv = 1.0f / red[0]; __syncthreads();
    for (int j = tid; j < L_SEQ; j += 128) sS[r * L_SEQ + j] *= inv;
  }
  __syncthreads();

  // ---- Stage 4: context = attn (16x2048) @ V (2048x32) --------------------
  for (int t = tid; t < 16 * D_HEAD; t += 128) sC[t] = 0.0f;
  __syncthreads();

  v8f acc0 = {}, acc1 = {};
  for (int kt = wave; kt < L_SEQ / 32; kt += 4) {
    // A from probs in LDS: two contiguous 8-float runs per lane -> ds_load_b128
    const float4* s4 = (const float4*)(sS + lr * L_SEQ + kt * 32 + (half << 3));
    const v8u aA = pk8(s4[0], s4[1], s4[4], s4[5]);
    // B from pre-packed V pairs: 8 direct u32 loads per operand
    v8u bV0, bV1;
#pragma unroll
    for (int v = 0; v < 8; ++v) {
      const int k2 = kt * 16 + (half << 2) + (v & 3) + ((v >> 2) << 3);
      const unsigned* vp = Vb + ((size_t)(bh * (L_SEQ / 2) + k2)) * 32;
      bV0[v] = vp[lr];
      bV1[v] = vp[16 + lr];
    }
    acc0 = wmma_pk(aA, bV0, acc0);
    acc1 = wmma_pk(aA, bV1, acc1);
  }
#pragma unroll
  for (int v = 0; v < 8; ++v) {
    const int r = v + 8 * half;
    atomicAdd(&sC[r * D_HEAD + lr],      acc0[v]);   // ds_add_f32
    atomicAdd(&sC[r * D_HEAD + 16 + lr], acc1[v]);
  }
  __syncthreads();

  // ---- Stage 5: write context block as packed bf16 pairs ------------------
  for (int t = tid; t < 16 * (D_HEAD / 2); t += 128) {
    const int r  = t >> 4;
    const int d2 = t & 15;
    ctxp[((size_t)(b * L_SEQ + i0 + r)) * (D_MODEL / 2) + h * (D_HEAD / 2) + d2] =
        pk_bf16(sC[r * D_HEAD + 2 * d2], sC[r * D_HEAD + 2 * d2 + 1]);
  }
}

// ---------------------------------------------------------------------------
// Host-side launch
// ---------------------------------------------------------------------------
extern "C" void kernel_launch(void* const* d_in, const int* in_sizes, int n_in,
                              void* d_out, int out_size, void* d_ws, size_t ws_size,
                              hipStream_t stream) {
  (void)in_sizes; (void)n_in; (void)out_size; (void)ws_size;

  const float* query = (const float*)d_in[0];
  const float* key   = (const float*)d_in[1];
  const float* value = (const float*)d_in[2];
  const float* pos   = (const float*)d_in[3];
  const float* Wq    = (const float*)d_in[4];
  const float* bq    = (const float*)d_in[5];
  const float* Wk    = (const float*)d_in[6];
  const float* bk    = (const float*)d_in[7];
  const float* Wv    = (const float*)d_in[8];
  const float* bv    = (const float*)d_in[9];
  const float* Wp    = (const float*)d_in[10];
  const float* ub    = (const float*)d_in[11];
  const float* vb    = (const float*)d_in[12];
  const float* Wo    = (const float*)d_in[13];
  const float* bo    = (const float*)d_in[14];
  float* out = (float*)d_out;

  const int BL = BATCH * L_SEQ;              // 4096
  const int RP = D_MODEL / 2;                // 256 u32 pairs per row
  float* Qf = (float*)d_ws;                  // BL x 512 f32
  float* Kf = Qf + (size_t)BL * D_MODEL;
  float* Vf = Kf + (size_t)BL * D_MODEL;
  float* Pf = Vf + (size_t)BL * D_MODEL;     // L x 512 f32
  unsigned* Kb  = (unsigned*)(Pf + (size_t)L_SEQ * D_MODEL);     // BH*L*16
  unsigned* Vb  = Kb + (size_t)BATCH * NUM_HEADS * L_SEQ * 16;   // BH*(L/2)*32
  unsigned* Pb  = Vb + (size_t)BATCH * NUM_HEADS * (L_SEQ / 2) * 32; // H*L*16
  unsigned* Xq  = Pb + (size_t)NUM_HEADS * L_SEQ * 16;           // BL*256
  unsigned* Xk  = Xq + (size_t)BL * RP;
  unsigned* Xv  = Xk + (size_t)BL * RP;
  unsigned* Xpp = Xv + (size_t)BL * RP;       // L*256
  unsigned* Cxp = Xpp + (size_t)L_SEQ * RP;   // BL*256 (packed context)
  unsigned* Wqp = Cxp + (size_t)BL * RP;      // 512*256 each
  unsigned* Wkp = Wqp + (size_t)D_MODEL * RP;
  unsigned* Wvp = Wkp + (size_t)D_MODEL * RP;
  unsigned* Wpp = Wvp + (size_t)D_MODEL * RP;
  unsigned* Wop = Wpp + (size_t)D_MODEL * RP;

  dim3 blk(256);
  // Pack all GEMM inputs to bf16 pairs (once)
  pack_pairs_bf16<<<dim3(BL * RP / 256), blk, 0, stream>>>(query, Xq, BL * RP);
  pack_pairs_bf16<<<dim3(BL * RP / 256), blk, 0, stream>>>(key,   Xk, BL * RP);
  pack_pairs_bf16<<<dim3(BL * RP / 256), blk, 0, stream>>>(value, Xv, BL * RP);
  pack_pairs_bf16<<<dim3(L_SEQ * RP / 256), blk, 0, stream>>>(pos, Xpp, L_SEQ * RP);
  pack_pairs_bf16<<<dim3(D_MODEL * RP / 256), blk, 0, stream>>>(Wq, Wqp, D_MODEL * RP);
  pack_pairs_bf16<<<dim3(D_MODEL * RP / 256), blk, 0, stream>>>(Wk, Wkp, D_MODEL * RP);
  pack_pairs_bf16<<<dim3(D_MODEL * RP / 256), blk, 0, stream>>>(Wv, Wvp, D_MODEL * RP);
  pack_pairs_bf16<<<dim3(D_MODEL * RP / 256), blk, 0, stream>>>(Wp, Wpp, D_MODEL * RP);
  pack_pairs_bf16<<<dim3(D_MODEL * RP / 256), blk, 0, stream>>>(Wo, Wop, D_MODEL * RP);

  // Projections
  gemm_bias_wmma<<<dim3(BL / 16, 4), blk, 0, stream>>>(Xq, Wqp, bq, Qf, BL);
  gemm_bias_wmma<<<dim3(BL / 16, 4), blk, 0, stream>>>(Xk, Wkp, bk, Kf, BL);
  gemm_bias_wmma<<<dim3(BL / 16, 4), blk, 0, stream>>>(Xv, Wvp, bv, Vf, BL);
  gemm_bias_wmma<<<dim3(L_SEQ / 16, 4), blk, 0, stream>>>(Xpp, Wpp, nullptr, Pf, L_SEQ);

  // Pre-pack K/P/V to bf16 in attention B-operand layouts
  pack_rows_bf16<<<dim3((BATCH * NUM_HEADS * L_SEQ * 16) / 256), blk, 0, stream>>>(Kf, Kb, BATCH);
  pack_rows_bf16<<<dim3((NUM_HEADS * L_SEQ * 16) / 256), blk, 0, stream>>>(Pf, Pb, 1);
  pack_vpairs_bf16<<<dim3((BATCH * NUM_HEADS * (L_SEQ / 2) * D_HEAD) / 256), blk, 0, stream>>>(Vf, Vb);

  // Relative attention core: one workgroup per (b*h, 16-row block)
  const size_t ldsBytes = (size_t)(17 * L_SEQ + 16 * L_SEQ + 16 * D_HEAD + 128) * sizeof(float);
  rel_attn_wmma<<<dim3(BATCH * NUM_HEADS, L_SEQ / 16), dim3(128), ldsBytes, stream>>>(
      Qf, Kb, Vb, Pb, ub, vb, Cxp);

  // Output projection (consumes packed context directly)
  gemm_bias_wmma<<<dim3(BL / 16, 4), blk, 0, stream>>>(Cxp, Wop, bo, out, BL);
}
